// CrossModalMoERouter_24515673326147
// MI455X (gfx1250) — matrix operand order repaired and checked by reading, not compiled, not run on hardware
//
#include <hip/hip_runtime.h>

typedef __attribute__((ext_vector_type(16))) _Float16 v16h;
typedef __attribute__((ext_vector_type(8)))  float    v8f;
typedef __attribute__((ext_vector_type(2)))  _Float16 h2v;

#define BTOK 8192
#define DDIM 1024
#define HDIM 4096
#define NEXP 4
#define KSEL 2
#define LN_EPS 1e-5f

#define BM 128
#define BN 128
#define BK 32
#define TPE (BTOK / BM)   // 64 tiles per expert (worst case)

// ---- workspace byte offsets ----
#define CNT_OFF   0                                  // NEXP ints
#define PSUM_OFF  64                                 // NEXP floats
#define MU_OFF    256                                // BTOK floats
#define RSTD_OFF  (MU_OFF   + BTOK * 4)              // BTOK floats
#define WSLOT_OFF (RSTD_OFF + BTOK * 4)              // BTOK*KSEL floats
#define ROUTE_OFF (WSLOT_OFF + BTOK * KSEL * 4)      // NEXP*BTOK ints (rowslot = row*2+slot)
#define HBUF_OFF  (ROUTE_OFF + NEXP * BTOK * 4 + 256) // BTOK*KSEL*HDIM f16 (~128 MB)

// ---------------- zero kernel ----------------
__global__ void zero_kernel(float4* __restrict__ y4, char* __restrict__ ws) {
  size_t i = (size_t)blockIdx.x * blockDim.x + threadIdx.x;
  if (i < (size_t)BTOK * DDIM / 4) y4[i] = make_float4(0.f, 0.f, 0.f, 0.f);
  if (blockIdx.x == 0 && threadIdx.x < NEXP) {
    ((int*)(ws + CNT_OFF))[threadIdx.x] = 0;
    ((float*)(ws + PSUM_OFF))[threadIdx.x] = 0.f;
  }
}

// ---------------- gating: LN stats + softmax + top-2 + routing ----------------
__global__ void gate_kernel(const float* __restrict__ x,
                            const float* __restrict__ gw,
                            const float* __restrict__ gb,
                            float* __restrict__ probsOut,
                            float* __restrict__ tidxOut,
                            float* __restrict__ twOut,
                            char* __restrict__ ws) {
  const int wave = threadIdx.x >> 5;
  const int lane = threadIdx.x & 31;
  const int row  = blockIdx.x * 8 + wave;
  const float* xr = x + (size_t)row * DDIM;

  float s = 0.f, s2 = 0.f, d0 = 0.f, d1 = 0.f, d2 = 0.f, d3 = 0.f;
  for (int i = lane; i < DDIM; i += 32) {
    float v = xr[i];
    s  += v;  s2 += v * v;
    d0 += v * gw[0 * DDIM + i];
    d1 += v * gw[1 * DDIM + i];
    d2 += v * gw[2 * DDIM + i];
    d3 += v * gw[3 * DDIM + i];
  }
  for (int off = 16; off > 0; off >>= 1) {
    s  += __shfl_xor(s,  off, 32);
    s2 += __shfl_xor(s2, off, 32);
    d0 += __shfl_xor(d0, off, 32);
    d1 += __shfl_xor(d1, off, 32);
    d2 += __shfl_xor(d2, off, 32);
    d3 += __shfl_xor(d3, off, 32);
  }
  if (lane == 0) {
    float mu   = s / (float)DDIM;
    float var  = s2 / (float)DDIM - mu * mu;
    float rstd = rsqrtf(var + LN_EPS);
    float lg[4] = { d0 + gb[0], d1 + gb[1], d2 + gb[2], d3 + gb[3] };
    float mx = fmaxf(fmaxf(lg[0], lg[1]), fmaxf(lg[2], lg[3]));
    float pe[4]; float psum = 0.f;
    for (int e = 0; e < 4; ++e) { pe[e] = expf(lg[e] - mx); psum += pe[e]; }
    for (int e = 0; e < 4; ++e) pe[e] /= psum;

    int i0 = 0;
    for (int e = 1; e < 4; ++e) if (pe[e] > pe[i0]) i0 = e;
    int i1 = -1;
    for (int e = 0; e < 4; ++e) { if (e == i0) continue; if (i1 < 0 || pe[e] > pe[i1]) i1 = e; }
    float w0 = pe[i0], w1 = pe[i1];
    float inv = 1.0f / (w0 + w1 + 1e-9f);
    w0 *= inv; w1 *= inv;

    for (int e = 0; e < 4; ++e) probsOut[row * 4 + e] = pe[e];
    tidxOut[row * 2 + 0] = (float)i0;
    tidxOut[row * 2 + 1] = (float)i1;
    twOut[row * 2 + 0] = w0;
    twOut[row * 2 + 1] = w1;

    ((float*)(ws + MU_OFF))[row]   = mu;
    ((float*)(ws + RSTD_OFF))[row] = rstd;
    float* wsl = (float*)(ws + WSLOT_OFF);
    wsl[row * 2 + 0] = w0;
    wsl[row * 2 + 1] = w1;

    int*   cnt   = (int*)(ws + CNT_OFF);
    int*   route = (int*)(ws + ROUTE_OFF);
    float* psm   = (float*)(ws + PSUM_OFF);
    int p0 = atomicAdd(&cnt[i0], 1); route[i0 * BTOK + p0] = row * 2 + 0;
    int p1 = atomicAdd(&cnt[i1], 1); route[i1 * BTOK + p1] = row * 2 + 1;
    for (int e = 0; e < 4; ++e) atomicAdd(&psm[e], pe[e]);
  }
}

// ---------------- aux loss finalize ----------------
__global__ void aux_kernel(char* __restrict__ ws, float* __restrict__ auxOut) {
  const int*   cnt = (const int*)(ws + CNT_OFF);
  const float* ps  = (const float*)(ws + PSUM_OFF);
  float s = 0.f;
  for (int e = 0; e < NEXP; ++e) {
    float frac = (float)cnt[e] / (float)(BTOK * KSEL);
    float mp   = ps[e] / (float)BTOK;
    s += frac * mp;
  }
  *auxOut = (float)NEXP * s;
}

// ---------------- GEMM1: h = GELU(xe @ w1 + b1), routed, f16 WMMA ----------------
__launch_bounds__(256, 1)
__global__ void moe_gemm1(const float* __restrict__ x,
                          const float* __restrict__ lnw,
                          const float* __restrict__ lnb,
                          const float* __restrict__ w1,
                          const float* __restrict__ b1,
                          char* __restrict__ ws) {
  __shared__ _Float16 As[2][BM * BK];
  __shared__ _Float16 Bs[2][BN * BK];
  __shared__ int   sSlot[BM];
  __shared__ float sMu[BM];
  __shared__ float sRs[BM];

  const int*   cnt   = (const int*)(ws + CNT_OFF);
  const int*   route = (const int*)(ws + ROUTE_OFF);
  const float* muA   = (const float*)(ws + MU_OFF);
  const float* rsA   = (const float*)(ws + RSTD_OFF);
  _Float16*    hbuf  = (_Float16*)(ws + HBUF_OFF);

  const int e     = blockIdx.x / TPE;
  const int m0    = (blockIdx.x % TPE) * BM;
  const int n0    = blockIdx.y * BN;
  const int count = cnt[e];
  if (m0 >= count) return;                 // uniform: EXEC stays all-1 for WMMA
  const int validM = min(BM, count - m0);

  const int tid = threadIdx.x;
  if (tid < BM) {
    int mi  = min(m0 + tid, count - 1);
    int rs  = route[e * BTOK + mi];
    sSlot[tid] = rs;
    int row = rs >> 1;
    sMu[tid] = muA[row];
    sRs[tid] = rsA[row];
  }
  __syncthreads();

  const float*  lnwE = lnw + (size_t)e * DDIM;
  const float*  lnbE = lnb + (size_t)e * DDIM;
  const float*  w1E  = w1  + (size_t)e * DDIM * HDIM + n0;
  const float2* x2   = (const float2*)x;

  // ---- hoisted, k-invariant gather state (32-bit element offsets) ----
  const int rBase = tid >> 4;          // 0..15 : A-row group
  const int kkA   = (tid & 15) * 2;    // fixed K-pair within chunk
  const int laA   = rBase + 16 * (kkA >> 4);
  const int jaA   = kkA & 15;          // even
  const int nB    = tid & 127;         // B column
  const int pqB   = tid >> 7;          // 0..1 : B K-pair group base
  const int nbB   = (nB >> 4) * 512;
  const int lnB16 = nB & 15;

  int   aOff[8];                       // float2-unit offsets into x
  float muV[8], rsV[8];
  #pragma unroll
  for (int i = 0; i < 8; ++i) {
    int r = rBase + 16 * i;
    aOff[i] = (sSlot[r] >> 1) * (DDIM / 2) + (kkA >> 1);
    muV[i] = sMu[r];
    rsV[i] = sRs[r];
  }

  float2 aV[8];
  float  bV0[8], bV1[8];
  float2 lwV, lbV;

  // stage chunk 0: global -> regs
  {
    lwV = *(const float2*)&lnwE[kkA];
    lbV = *(const float2*)&lnbE[kkA];
    #pragma unroll
    for (int i = 0; i < 8; ++i) {
      aV[i] = x2[aOff[i]];
      int kb0 = 2 * (pqB + 2 * i);
      bV0[i] = w1E[(size_t)kb0 * HDIM + nB];
      bV1[i] = w1E[(size_t)(kb0 + 1) * HDIM + nB];
    }
    // regs -> LDS (fragment order, paired 32-bit stores)
    #pragma unroll
    for (int i = 0; i < 8; ++i) {
      float v0 = (aV[i].x - muV[i]) * rsV[i] * lwV.x + lbV.x;
      float v1 = (aV[i].y - muV[i]) * rsV[i] * lwV.y + lbV.y;
      *(h2v*)&As[0][i * 512 + laA * 16 + jaA] = (h2v){(_Float16)v0, (_Float16)v1};
      int kb0 = 2 * (pqB + 2 * i);
      int idxB = nbB + (lnB16 + 16 * (kb0 >> 4)) * 16 + (kb0 & 15);
      *(h2v*)&Bs[0][idxB] = (h2v){(_Float16)bV0[i], (_Float16)bV1[i]};
    }
  }
  __syncthreads();

  const int lane = tid & 31;
  const int wm = (tid >> 5) >> 1;   // 0..3
  const int wn = (tid >> 5) & 1;    // 0..1

  v8f acc[2][4];
  #pragma unroll
  for (int i = 0; i < 2; ++i)
    #pragma unroll
    for (int j = 0; j < 4; ++j) acc[i][j] = (v8f)(0.0f);

  const int KT = DDIM / BK;  // 32
  for (int kt = 0; kt < KT; ++kt) {
    const int  buf  = kt & 1;
    const bool more = (kt + 1) < KT;
    if (more) {
      int k0 = (kt + 1) * BK;
      lwV = *(const float2*)&lnwE[k0 + kkA];
      lbV = *(const float2*)&lnbE[k0 + kkA];
      #pragma unroll
      for (int i = 0; i < 8; ++i) {
        aV[i] = x2[aOff[i] + (k0 >> 1)];
        int kb0 = 2 * (pqB + 2 * i);
        bV0[i] = w1E[(size_t)(k0 + kb0) * HDIM + nB];
        bV1[i] = w1E[(size_t)(k0 + kb0 + 1) * HDIM + nB];
      }
    }
    v16h af[2], bf[4];
    #pragma unroll
    for (int i = 0; i < 2; ++i)
      af[i] = *(const v16h*)&As[buf][((wm * 2 + i) * 32 + lane) * 16];
    #pragma unroll
    for (int j = 0; j < 4; ++j)
      bf[j] = *(const v16h*)&Bs[buf][((wn * 4 + j) * 32 + lane) * 16];
    #pragma unroll
    for (int i = 0; i < 2; ++i)
      #pragma unroll
      for (int j = 0; j < 4; ++j)
        acc[i][j] = __builtin_amdgcn_wmma_f32_16x16x32_f16(
            false, af[i], false, bf[j], (short)0, acc[i][j], false, false);
    if (more) {
      __syncthreads();
      int nb = buf ^ 1;
      #pragma unroll
      for (int i = 0; i < 8; ++i) {
        float v0 = (aV[i].x - muV[i]) * rsV[i] * lwV.x + lbV.x;
        float v1 = (aV[i].y - muV[i]) * rsV[i] * lwV.y + lbV.y;
        *(h2v*)&As[nb][i * 512 + laA * 16 + jaA] = (h2v){(_Float16)v0, (_Float16)v1};
        int kb0 = 2 * (pqB + 2 * i);
        int idxB = nbB + (lnB16 + 16 * (kb0 >> 4)) * 16 + (kb0 & 15);
        *(h2v*)&Bs[nb][idxB] = (h2v){(_Float16)bV0[i], (_Float16)bV1[i]};
      }
      __syncthreads();
    }
  }

  // epilogue: bias + exact GELU -> f16 h
  const int nlo = lane & 15;
  const int mhi = (lane >> 4) * 8;
  #pragma unroll
  for (int i = 0; i < 2; ++i) {
    #pragma unroll
    for (int j = 0; j < 4; ++j) {
      int   nCol = n0 + wn * 64 + j * 16 + nlo;
      float bb   = b1[(size_t)e * HDIM + nCol];
      #pragma unroll
      for (int r = 0; r < 8; ++r) {
        int mLoc = wm * 32 + i * 16 + mhi + r;
        if (mLoc < validM) {
          float v = acc[i][j][r] + bb;
          v = 0.5f * v * (1.0f + erff(v * 0.70710678118654752f));
          hbuf[(unsigned)(sSlot[mLoc] * HDIM + nCol)] = (_Float16)v;
        }
      }
    }
  }
}

// ---------------- GEMM2: y += gate_w * (h @ w2 + b2), routed, f16 WMMA ----------------
__launch_bounds__(256, 1)
__global__ void moe_gemm2(const float* __restrict__ w2,
                          const float* __restrict__ b2,
                          char* __restrict__ ws,
                          float* __restrict__ y) {
  __shared__ _Float16 As[2][BM * BK];
  __shared__ _Float16 Bs[2][BN * BK];
  __shared__ int   sSlot[BM];
  __shared__ float sWt[BM];

  const int*      cnt   = (const int*)(ws + CNT_OFF);
  const int*      route = (const int*)(ws + ROUTE_OFF);
  const float*    wsl   = (const float*)(ws + WSLOT_OFF);
  const h2v*      h2p   = (const h2v*)(ws + HBUF_OFF);

  const int e     = blockIdx.x / TPE;
  const int m0    = (blockIdx.x % TPE) * BM;
  const int n0    = blockIdx.y * BN;
  const int count = cnt[e];
  if (m0 >= count) return;
  const int validM = min(BM, count - m0);

  const int tid = threadIdx.x;
  if (tid < BM) {
    int mi = min(m0 + tid, count - 1);
    int rs = route[e * BTOK + mi];
    sSlot[tid] = rs;
    sWt[tid]   = wsl[rs];
  }
  __syncthreads();

  const float* w2E = w2 + (size_t)e * HDIM * DDIM + n0;

  const int rBase = tid >> 4;
  const int kkA   = (tid & 15) * 2;
  const int laA   = rBase + 16 * (kkA >> 4);
  const int jaA   = kkA & 15;
  const int nB    = tid & 127;
  const int pqB   = tid >> 7;
  const int nbB   = (nB >> 4) * 512;
  const int lnB16 = nB & 15;

  int aOff[8];                         // h2-unit offsets into hbuf
  #pragma unroll
  for (int i = 0; i < 8; ++i)
    aOff[i] = sSlot[rBase + 16 * i] * (HDIM / 2) + (kkA >> 1);

  h2v   aV[8];
  float bV0[8], bV1[8];

  {
    #pragma unroll
    for (int i = 0; i < 8; ++i) {
      aV[i] = h2p[aOff[i]];
      int kb0 = 2 * (pqB + 2 * i);
      bV0[i] = w2E[(size_t)kb0 * DDIM + nB];
      bV1[i] = w2E[(size_t)(kb0 + 1) * DDIM + nB];
    }
    #pragma unroll
    for (int i = 0; i < 8; ++i) {
      *(h2v*)&As[0][i * 512 + laA * 16 + jaA] = aV[i];
      int kb0 = 2 * (pqB + 2 * i);
      int idxB = nbB + (lnB16 + 16 * (kb0 >> 4)) * 16 + (kb0 & 15);
      *(h2v*)&Bs[0][idxB] = (h2v){(_Float16)bV0[i], (_Float16)bV1[i]};
    }
  }
  __syncthreads();

  const int lane = tid & 31;
  const int wm = (tid >> 5) >> 1;
  const int wn = (tid >> 5) & 1;

  v8f acc[2][4];
  #pragma unroll
  for (int i = 0; i < 2; ++i)
    #pragma unroll
    for (int j = 0; j < 4; ++j) acc[i][j] = (v8f)(0.0f);

  const int KT = HDIM / BK;  // 128
  for (int kt = 0; kt < KT; ++kt) {
    const int  buf  = kt & 1;
    const bool more = (kt + 1) < KT;
    if (more) {
      int k0 = (kt + 1) * BK;
      #pragma unroll
      for (int i = 0; i < 8; ++i) {
        aV[i] = h2p[aOff[i] + (k0 >> 1)];
        int kb0 = 2 * (pqB + 2 * i);
        bV0[i] = w2E[(size_t)(k0 + kb0) * DDIM + nB];
        bV1[i] = w2E[(size_t)(k0 + kb0 + 1) * DDIM + nB];
      }
    }
    v16h af[2], bf[4];
    #pragma unroll
    for (int i = 0; i < 2; ++i)
      af[i] = *(const v16h*)&As[buf][((wm * 2 + i) * 32 + lane) * 16];
    #pragma unroll
    for (int j = 0; j < 4; ++j)
      bf[j] = *(const v16h*)&Bs[buf][((wn * 4 + j) * 32 + lane) * 16];
    #pragma unroll
    for (int i = 0; i < 2; ++i)
      #pragma unroll
      for (int j = 0; j < 4; ++j)
        acc[i][j] = __builtin_amdgcn_wmma_f32_16x16x32_f16(
            false, af[i], false, bf[j], (short)0, acc[i][j], false, false);
    if (more) {
      __syncthreads();
      int nb = buf ^ 1;
      #pragma unroll
      for (int i = 0; i < 8; ++i) {
        *(h2v*)&As[nb][i * 512 + laA * 16 + jaA] = aV[i];
        int kb0 = 2 * (pqB + 2 * i);
        int idxB = nbB + (lnB16 + 16 * (kb0 >> 4)) * 16 + (kb0 & 15);
        *(h2v*)&Bs[nb][idxB] = (h2v){(_Float16)bV0[i], (_Float16)bV1[i]};
      }
      __syncthreads();
    }
  }

  // epilogue: bias, gate-weight scale, accumulate into y (2 commutative adds/elem)
  const int nlo = lane & 15;
  const int mhi = (lane >> 4) * 8;
  #pragma unroll
  for (int i = 0; i < 2; ++i) {
    #pragma unroll
    for (int j = 0; j < 4; ++j) {
      int   nCol = n0 + wn * 64 + j * 16 + nlo;
      float bb   = b2[(size_t)e * DDIM + nCol];
      #pragma unroll
      for (int r = 0; r < 8; ++r) {
        int mLoc = wm * 32 + i * 16 + mhi + r;
        if (mLoc < validM) {
          int   rs = sSlot[mLoc];
          float v  = (acc[i][j][r] + bb) * sWt[mLoc];
          atomicAdd(&y[(unsigned)((rs >> 1) * DDIM + nCol)], v);
        }
      }
    }
  }
}

// ---------------- launch ----------------
extern "C" void kernel_launch(void* const* d_in, const int* in_sizes, int n_in,
                              void* d_out, int out_size, void* d_ws, size_t ws_size,
                              hipStream_t stream) {
  const float* x   = (const float*)d_in[0];
  const float* gw  = (const float*)d_in[1];
  const float* gb  = (const float*)d_in[2];
  const float* lnw = (const float*)d_in[3];
  const float* lnb = (const float*)d_in[4];
  const float* w1  = (const float*)d_in[5];
  const float* b1  = (const float*)d_in[6];
  const float* w2  = (const float*)d_in[7];
  const float* b2  = (const float*)d_in[8];
  (void)in_sizes; (void)n_in; (void)out_size; (void)ws_size;

  float* out   = (float*)d_out;
  char*  ws    = (char*)d_ws;
  float* y     = out;
  float* probs = out + (size_t)BTOK * DDIM;
  float* tidx  = probs + (size_t)BTOK * NEXP;
  float* tw    = tidx + (size_t)BTOK * KSEL;
  float* aux   = tw + (size_t)BTOK * KSEL;

  zero_kernel<<<(BTOK * DDIM / 4 + 255) / 256, 256, 0, stream>>>((float4*)y, ws);
  gate_kernel<<<BTOK / 8, 256, 0, stream>>>(x, gw, gb, probs, tidx, tw, ws);
  aux_kernel<<<1, 1, 0, stream>>>(ws, aux);
  moe_gemm1<<<dim3(NEXP * TPE, HDIM / BN), 256, 0, stream>>>(x, lnw, lnb, w1, b1, ws);
  moe_gemm2<<<dim3(NEXP * TPE, DDIM / BN), 256, 0, stream>>>(w2, b2, ws, y);
}
